// ThoughtEngine_82626580841048
// MI455X (gfx1250) — compile-verified, matrix operand order checked by reading, not run on hardware
//
#include <hip/hip_runtime.h>
#include <hip/hip_bf16.h>

#define B_  8
#define L_  1024
#define D_  1024
#define S_  32
#define C_  8
#define H_  16
#define HD_ 64

typedef __bf16 bf16x16 __attribute__((ext_vector_type(16)));
typedef float  f32x8   __attribute__((ext_vector_type(8)));
typedef unsigned int u32x4 __attribute__((ext_vector_type(4)));
typedef int  i32x4 __attribute__((ext_vector_type(4)));
typedef int  i32x8 __attribute__((ext_vector_type(8)));

union FragU { uint4 u[2]; bf16x16 v; };

// Use the Tensor Data Mover when the toolchain exposes it (device pass only).
#if defined(__has_builtin)
#if __has_builtin(__builtin_amdgcn_tensor_load_to_lds) && \
    __has_builtin(__builtin_amdgcn_s_wait_tensorcnt)
#define USE_TDM 1
#endif
#endif

// ---------------------------------------------------------------------------
// Reductions (wave32)
// ---------------------------------------------------------------------------
__device__ __forceinline__ float waveSum(float v) {
#pragma unroll
  for (int o = 16; o > 0; o >>= 1) v += __shfl_xor(v, o, 32);
  return v;
}
__device__ __forceinline__ float waveMax(float v) {
#pragma unroll
  for (int o = 16; o > 0; o >>= 1) v = fmaxf(v, __shfl_xor(v, o, 32));
  return v;
}
__device__ __forceinline__ float blockSum(float v, float* red) {
  const int lane = threadIdx.x & 31, w = threadIdx.x >> 5;
  const int nw = blockDim.x >> 5;
  v = waveSum(v);
  if (lane == 0) red[w] = v;
  __syncthreads();
  float r = (threadIdx.x < nw) ? red[threadIdx.x] : 0.0f;
  if (w == 0) { r = waveSum(r); if (lane == 0) red[0] = r; }
  __syncthreads();
  float out = red[0];
  __syncthreads();
  return out;
}
__device__ __forceinline__ float blockMax(float v, float* red) {
  const int lane = threadIdx.x & 31, w = threadIdx.x >> 5;
  const int nw = blockDim.x >> 5;
  v = waveMax(v);
  if (lane == 0) red[w] = v;
  __syncthreads();
  float r = (threadIdx.x < nw) ? red[threadIdx.x] : -1e30f;
  if (w == 0) { r = waveMax(r); if (lane == 0) red[0] = r; }
  __syncthreads();
  float out = red[0];
  __syncthreads();
  return out;
}

__device__ __forceinline__ void ln1024(float x, float* out_arr,
                                       const float* g, const float* bt,
                                       float* red) {
  float m = blockSum(x, red) * (1.0f / 1024.0f);
  float d = x - m;
  float v = blockSum(d * d, red) * (1.0f / 1024.0f);
  out_arr[threadIdx.x] = d * rsqrtf(v + 1e-5f) * g[threadIdx.x] + bt[threadIdx.x];
  __syncthreads();
}

// ---------------------------------------------------------------------------
// fp32 -> bf16 conversion
// ---------------------------------------------------------------------------
__global__ void cvt_bf16_kernel(const float* __restrict__ in,
                                __hip_bfloat16* __restrict__ out, int n) {
  int i = blockIdx.x * 256 + threadIdx.x;
  if (i < n) out[i] = __float2bfloat16(in[i]);
}

#ifdef USE_TDM
// ---------------------------------------------------------------------------
// TDM: DMA one 128-row x 32-col bf16 tile (rows of 64B) from a row-major
// matrix with row stride `rowStrideElems` into LDS at byte offset lds_off.
// LDS padding: after every 16 DWORDs (64B row) insert 4 DWORDs (16B) ->
// 40-half LDS row stride (matches the compute-side layout, conflict-free).
// Descriptor layout per CDNA5 ISA.
// ---------------------------------------------------------------------------
__device__ __forceinline__ void tdm_load_tile_128x32(uint32_t lds_off,
                                                     const __hip_bfloat16* gtile,
                                                     int rowStrideElems) {
  uint64_t ga = (uint64_t)(uintptr_t)gtile;
  u32x4 g0;
  g0[0] = 1u;                                   // count=1, user mode, no gather
  g0[1] = lds_off;                              // LDS byte address
  g0[2] = (uint32_t)ga;                         // global_addr[31:0]
  g0[3] = (uint32_t)(ga >> 32) | (2u << 30);    // global_addr[56:32] | type=2

  uint32_t K = (uint32_t)rowStrideElems;
  i32x8 g1;
  g1[0] = (int)((1u << 16) |   // data_size = 2 bytes
                (1u << 20) |   // pad_enable
                (3u << 22) |   // pad_interval: 16 DWORDs
                (3u << 25));   // pad_amount:  4 DWORDs
  g1[1] = (int)((K & 0xFFFFu) << 16);                    // tensor_dim0[15:0]
  g1[2] = (int)(((K >> 16) & 0xFFFFu) | 0xFFFF0000u);    // dim0[31:16] | dim1 lo
  g1[3] = (int)(0x0000FFFFu | (32u << 16));              // dim1 hi | tile_dim0=32
  g1[4] = (int)128u;                                     // tile_dim1=128, tile_dim2=0
  g1[5] = (int)K;                                        // tensor_dim0_stride[31:0]
  g1[6] = 0;                                             // stride hi | dim1_stride lo
  g1[7] = 0;                                             // dim1_stride hi
  i32x4 gz = {0, 0, 0, 0};
#if defined(__clang_major__) && (__clang_major__ >= 23)
  i32x8 gz8 = {0, 0, 0, 0, 0, 0, 0, 0};
  __builtin_amdgcn_tensor_load_to_lds(g0, g1, gz, gz, gz8, 0);
#else
  __builtin_amdgcn_tensor_load_to_lds(g0, g1, gz, gz, 0);
#endif
}
#endif  // USE_TDM

// ---------------------------------------------------------------------------
// bf16 WMMA GEMM:  C[M x N] = A[M x K] @ W[N x K]^T + bias
// 128x128 block tile, K-step 32, 8 waves (2M x 4N), wave tile 64x32.
// Staging: TDM double-buffered DMA (preferred) or manual b128 loads.
// Output: fp32 (Cf) or bf16 (Cb). M, N multiples of 128; K multiple of 32.
// ---------------------------------------------------------------------------
__global__ __launch_bounds__(256)
void gemm_bf16_wmma(const __hip_bfloat16* __restrict__ A,
                    const __hip_bfloat16* __restrict__ W,
                    const float* __restrict__ bias,
                    float* __restrict__ Cf,
                    __hip_bfloat16* __restrict__ Cb,
                    int K, int ldc) {
  __shared__ __hip_bfloat16 As[2][128][40];  // 40-half rows: 16B aligned, conflict-free
  __shared__ __hip_bfloat16 Bs[2][128][40];
  const int bm = blockIdx.x * 128;
  const int bn = blockIdx.y * 128;
  const int tid  = threadIdx.x;
  const int wave = tid >> 5;
  const int lane = tid & 31;
  const int mw = (wave & 1) * 64;
  const int nw = (wave >> 1) * 32;
  const int fr = lane & 15;
  const int fk = (lane >> 4) * 8;

  f32x8 acc[4][2] = {};
  const int niter = K >> 5;

#ifdef USE_TDM
  const uint32_t aoff[2] = {(uint32_t)(uintptr_t)&As[0][0][0],
                            (uint32_t)(uintptr_t)&As[1][0][0]};
  const uint32_t boff[2] = {(uint32_t)(uintptr_t)&Bs[0][0][0],
                            (uint32_t)(uintptr_t)&Bs[1][0][0]};
  // Prologue: DMA first tile pair into buffer 0. TDM ignores EXEC, so keep it
  // behind a real branch: waves with EXEC==0 branch over and never issue.
  if (tid == 0) {
    tdm_load_tile_128x32(aoff[0], A + (size_t)bm * K, K);
    tdm_load_tile_128x32(boff[0], W + (size_t)bn * K, K);
    __builtin_amdgcn_s_wait_tensorcnt(0);
  }
  __syncthreads();

  for (int it = 0; it < niter; ++it) {
    const int cur = it & 1;
    // Kick next tile pair into the other buffer while we compute this one.
    if (tid == 0 && (it + 1) < niter) {
      const int k0n = (it + 1) * 32;
      tdm_load_tile_128x32(aoff[cur ^ 1], A + (size_t)bm * K + k0n, K);
      tdm_load_tile_128x32(boff[cur ^ 1], W + (size_t)bn * K + k0n, K);
    }

    FragU af[4], bf[2];
#pragma unroll
    for (int n = 0; n < 2; ++n) {
      bf[n].u[0] = *(const uint4*)(&Bs[cur][nw + n * 16 + fr][fk]);
      bf[n].u[1] = *(const uint4*)(&Bs[cur][nw + n * 16 + fr][fk + 16]);
    }
#pragma unroll
    for (int m = 0; m < 4; ++m) {
      af[m].u[0] = *(const uint4*)(&As[cur][mw + m * 16 + fr][fk]);
      af[m].u[1] = *(const uint4*)(&As[cur][mw + m * 16 + fr][fk + 16]);
    }
#pragma unroll
    for (int m = 0; m < 4; ++m)
#pragma unroll
      for (int n = 0; n < 2; ++n)
        acc[m][n] = __builtin_amdgcn_wmma_f32_16x16x32_bf16(
            false, af[m].v, false, bf[n].v, (short)0, acc[m][n], false, false);

    // Wave 0 waits for the next tiles' DMA, then everyone syncs (also protects
    // the buffer we just read from next iteration's DMA).
    if (tid == 0) __builtin_amdgcn_s_wait_tensorcnt(0);
    __syncthreads();
  }
#else
  for (int it = 0; it < niter; ++it) {
    const int k0 = it * 32;
#pragma unroll
    for (int c = 0; c < 2; ++c) {
      int idx = tid + c * 256;       // 512 chunks of 8 halves per matrix
      int r   = idx >> 2;            // row 0..127
      int cc  = (idx & 3) * 8;       // half col 0..24
      *(uint4*)(&As[0][r][cc]) = *(const uint4*)(A + (size_t)(bm + r) * K + k0 + cc);
      *(uint4*)(&Bs[0][r][cc]) = *(const uint4*)(W + (size_t)(bn + r) * K + k0 + cc);
    }
    __syncthreads();

    FragU af[4], bf[2];
#pragma unroll
    for (int n = 0; n < 2; ++n) {
      bf[n].u[0] = *(const uint4*)(&Bs[0][nw + n * 16 + fr][fk]);
      bf[n].u[1] = *(const uint4*)(&Bs[0][nw + n * 16 + fr][fk + 16]);
    }
#pragma unroll
    for (int m = 0; m < 4; ++m) {
      af[m].u[0] = *(const uint4*)(&As[0][mw + m * 16 + fr][fk]);
      af[m].u[1] = *(const uint4*)(&As[0][mw + m * 16 + fr][fk + 16]);
    }
#pragma unroll
    for (int m = 0; m < 4; ++m)
#pragma unroll
      for (int n = 0; n < 2; ++n)
        acc[m][n] = __builtin_amdgcn_wmma_f32_16x16x32_bf16(
            false, af[m].v, false, bf[n].v, (short)0, acc[m][n], false, false);
    __syncthreads();
  }
#endif

  const int rofs = (lane >> 4) * 8;
  const int ncol = lane & 15;
#pragma unroll
  for (int m = 0; m < 4; ++m) {
#pragma unroll
    for (int n = 0; n < 2; ++n) {
      int gn = bn + nw + n * 16 + ncol;
      float bv = bias ? bias[gn] : 0.0f;
#pragma unroll
      for (int r = 0; r < 8; ++r) {
        int gm = bm + mw + m * 16 + r + rofs;
        float v = acc[m][n][r] + bv;
        if (Cf) Cf[(size_t)gm * ldc + gn] = v;
        else    Cb[(size_t)gm * ldc + gn] = __float2bfloat16(v);
      }
    }
  }
}

// ---------------------------------------------------------------------------
// ctx = mean_L(hidden); tok0 = ctx @ tpw^T + tpb; buf[:,0] = tok0;
// kbuf/vbuf row 0 = K/V projection of tok0.   One block per batch.
// ---------------------------------------------------------------------------
__global__ __launch_bounds__(1024)
void init_tok_kernel(const float* __restrict__ hidden,
                     const float* __restrict__ tpw, const float* __restrict__ tpb,
                     const __hip_bfloat16* __restrict__ w_in,
                     const float* __restrict__ b_in,
                     float* __restrict__ tok_g, float* __restrict__ buf_g,
                     float* __restrict__ kbuf, float* __restrict__ vbuf) {
  const int b = blockIdx.x, tid = threadIdx.x;
  __shared__ float ctx_s[D_], tok_s[D_];
  const float* hb = hidden + (size_t)b * L_ * D_;
  float s = 0.0f;
  for (int l = 0; l < L_; ++l) s += hb[(size_t)l * D_ + tid];
  ctx_s[tid] = s * (1.0f / (float)L_);
  __syncthreads();
  {
    const float* wr = tpw + (size_t)tid * D_;
    float t = tpb[tid];
    for (int k = 0; k < D_; ++k) t += ctx_s[k] * wr[k];
    tok_s[tid] = t;
    tok_g[(size_t)b * D_ + tid] = t;
    buf_g[(size_t)b * S_ * D_ + tid] = t;
  }
  __syncthreads();
  {
    const __hip_bfloat16* wk = w_in + (size_t)(D_ + tid) * D_;
    const __hip_bfloat16* wv = w_in + (size_t)(2 * D_ + tid) * D_;
    float ak = b_in[D_ + tid], av = b_in[2 * D_ + tid];
    for (int k = 0; k < D_; ++k) {
      float x = tok_s[k];
      ak += x * __bfloat162float(wk[k]);
      av += x * __bfloat162float(wv[k]);
    }
    kbuf[(size_t)b * S_ * D_ + tid] = ak;
    vbuf[(size_t)b * S_ * D_ + tid] = av;
  }
}

// ---------------------------------------------------------------------------
// Sequential thought loop: one block per batch, 1024 threads, 31 steps.
// qkv: [B*L][3D] bf16 (Q at +0, K at +D, V at +2D).
// ---------------------------------------------------------------------------
__global__ __launch_bounds__(1024)
void loop_kernel(const __hip_bfloat16* __restrict__ qkv,
                 const __hip_bfloat16* __restrict__ w_in, const float* __restrict__ b_in,
                 const __hip_bfloat16* __restrict__ w_out, const float* __restrict__ b_out,
                 const __hip_bfloat16* __restrict__ w1, const float* __restrict__ b1,
                 const __hip_bfloat16* __restrict__ w2, const float* __restrict__ b2,
                 const float* __restrict__ ln1g, const float* __restrict__ ln1b,
                 const float* __restrict__ ln2g, const float* __restrict__ ln2b,
                 const float* __restrict__ tok_g,
                 float* __restrict__ buf_g,
                 float* __restrict__ kbuf, float* __restrict__ vbuf) {
  const int b = blockIdx.x, tid = threadIdx.x;
  __shared__ float tok_s[D_], q_s[D_], av_s[D_], sc_s[L_ + S_], ff_s[2 * D_];
  __shared__ float part_s[64 * 16], red_s[32];

  tok_s[tid] = tok_g[(size_t)b * D_ + tid];
  __syncthreads();

  for (int i = 1; i < S_; ++i) {
    const int nk = L_ + i;  // valid keys: hidden + buf rows 0..i-1

    // ---- Q projection ----
    {
      const __hip_bfloat16* wr = w_in + (size_t)tid * D_;
      float acc = b_in[tid];
      for (int k = 0; k < D_; ++k) acc += tok_s[k] * __bfloat162float(wr[k]);
      q_s[tid] = acc;
    }
    __syncthreads();

    // ---- per-head attention over nk keys ----
    for (int h = 0; h < H_; ++h) {
      const int qo = h * HD_;
      for (int j = tid; j < nk; j += 1024) {
        float s = 0.0f;
        if (j < L_) {
          const __hip_bfloat16* kr = qkv + ((size_t)(b * L_ + j)) * (3 * D_) + D_ + qo;
          for (int d = 0; d < HD_; ++d) s += q_s[qo + d] * __bfloat162float(kr[d]);
        } else {
          const float* kr = kbuf + ((size_t)(b * S_ + (j - L_))) * D_ + qo;
          for (int d = 0; d < HD_; ++d) s += q_s[qo + d] * kr[d];
        }
        sc_s[j] = s * 0.125f;  // 1/sqrt(64)
      }
      __syncthreads();

      float mx = -1e30f;
      for (int j = tid; j < nk; j += 1024) mx = fmaxf(mx, sc_s[j]);
      mx = blockMax(mx, red_s);
      float sum = 0.0f;
      for (int j = tid; j < nk; j += 1024) {
        float e = __expf(sc_s[j] - mx);
        sc_s[j] = e;
        sum += e;
      }
      sum = blockSum(sum, red_s);
      const float inv = 1.0f / sum;

      const int d = tid & 63, pp = tid >> 6;  // 16 partials per dim
      float acc = 0.0f;
      for (int j = pp; j < nk; j += 16) {
        float a = sc_s[j];
        float v;
        if (j < L_) v = __bfloat162float(qkv[((size_t)(b * L_ + j)) * (3 * D_) + 2 * D_ + qo + d]);
        else        v = vbuf[((size_t)(b * S_ + (j - L_))) * D_ + qo + d];
        acc += a * v;
      }
      part_s[d * 16 + pp] = acc;
      __syncthreads();
      if (tid < 64) {
        float s2 = 0.0f;
        for (int p2 = 0; p2 < 16; ++p2) s2 += part_s[tid * 16 + p2];
        av_s[qo + tid] = s2 * inv;
      }
      __syncthreads();
    }

    // ---- output projection + residual + LN1 ----
    {
      const __hip_bfloat16* wr = w_out + (size_t)tid * D_;
      float acc = b_out[tid];
      for (int k = 0; k < D_; ++k) acc += av_s[k] * __bfloat162float(wr[k]);
      float x = tok_s[tid] + acc;
      ln1024(x, tok_s, ln1g, ln1b, red_s);
    }

    // ---- FFN (exact gelu) + residual + LN2 ----
    for (int e = tid; e < 2 * D_; e += 1024) {
      const __hip_bfloat16* wr = w1 + (size_t)e * D_;
      float acc = b1[e];
      for (int k = 0; k < D_; ++k) acc += tok_s[k] * __bfloat162float(wr[k]);
      ff_s[e] = 0.5f * acc * (1.0f + erff(acc * 0.70710678118654752f));
    }
    __syncthreads();
    {
      const __hip_bfloat16* wr = w2 + (size_t)tid * (2 * D_);
      float acc = b2[tid];
      for (int k = 0; k < 2 * D_; ++k) acc += ff_s[k] * __bfloat162float(wr[k]);
      float x = tok_s[tid] + acc;
      ln1024(x, tok_s, ln2g, ln2b, red_s);
    }

    // ---- emit buf row i and its K/V ----
    buf_g[((size_t)b * S_ + i) * D_ + tid] = tok_s[tid];
    {
      const __hip_bfloat16* wk = w_in + (size_t)(D_ + tid) * D_;
      const __hip_bfloat16* wv = w_in + (size_t)(2 * D_ + tid) * D_;
      float ak = b_in[D_ + tid], av = b_in[2 * D_ + tid];
      for (int k = 0; k < D_; ++k) {
        float x = tok_s[k];
        ak += x * __bfloat162float(wk[k]);
        av += x * __bfloat162float(wv[k]);
      }
      kbuf[((size_t)b * S_ + i) * D_ + tid] = ak;
      vbuf[((size_t)b * S_ + i) * D_ + tid] = av;
    }
    __syncthreads();
  }
}

// ---------------------------------------------------------------------------
// Compressed summary: one block per (b,c).  Writes summ to output and K/V of
// summ rows into Ks/Vs.
// ---------------------------------------------------------------------------
__global__ __launch_bounds__(1024)
void comp_kernel(const float* __restrict__ comp_q, const float* __restrict__ comp_w,
                 const float* __restrict__ comp_b,
                 const float* __restrict__ buf_g,
                 const __hip_bfloat16* __restrict__ w_in, const float* __restrict__ b_in,
                 float* __restrict__ summ_out,
                 float* __restrict__ Ks, float* __restrict__ Vs) {
  const int bc = blockIdx.x, b = bc >> 3, c = bc & 7;
  const int tid = threadIdx.x, w = tid >> 5, lane = tid & 31;
  __shared__ float cq[D_], a_s[S_], ctx_s[D_], row_s[D_];

  cq[tid] = comp_q[(size_t)c * D_ + tid];
  __syncthreads();

  {  // warp w computes score for thought row s=w
    const float* tr = buf_g + ((size_t)b * S_ + w) * D_;
    float p = 0.0f;
    for (int k = lane; k < D_; k += 32) p += cq[k] * tr[k];
    p = waveSum(p);
    if (lane == 0) a_s[w] = p * (1.0f / 32.0f);  // 1/sqrt(1024)
  }
  __syncthreads();
  if (w == 0) {  // softmax over 32 in wave 0
    float v = a_s[lane];
    float mx = waveMax(v);
    float e = __expf(v - mx);
    float sm = waveSum(e);
    a_s[lane] = e / sm;
  }
  __syncthreads();
  {
    float acc = 0.0f;
    for (int s = 0; s < S_; ++s) acc += a_s[s] * buf_g[((size_t)b * S_ + s) * D_ + tid];
    ctx_s[tid] = acc;
  }
  __syncthreads();
  {
    const float* wr = comp_w + (size_t)tid * D_;
    float r = comp_b[tid];
    for (int k = 0; k < D_; ++k) r += ctx_s[k] * wr[k];
    row_s[tid] = r;
    summ_out[(size_t)bc * D_ + tid] = r;
  }
  __syncthreads();
  {
    const __hip_bfloat16* wk = w_in + (size_t)(D_ + tid) * D_;
    const __hip_bfloat16* wv = w_in + (size_t)(2 * D_ + tid) * D_;
    float ak = b_in[D_ + tid], av = b_in[2 * D_ + tid];
    for (int k = 0; k < D_; ++k) {
      float x = row_s[k];
      ak += x * __bfloat162float(wk[k]);
      av += x * __bfloat162float(wv[k]);
    }
    Ks[(size_t)bc * D_ + tid] = ak;
    Vs[(size_t)bc * D_ + tid] = av;
  }
}

// ---------------------------------------------------------------------------
// Final MHA: every (b,l,h) attends over C=8 summ keys. One thread per (b,l,h).
// Writes bf16 o (feeds the WMMA out-projection GEMM).
// ---------------------------------------------------------------------------
__global__ __launch_bounds__(256)
void final_attn_kernel(const __hip_bfloat16* __restrict__ qkv,
                       const float* __restrict__ Ks, const float* __restrict__ Vs,
                       __hip_bfloat16* __restrict__ o_bf) {
  const int idx = blockIdx.x * 256 + threadIdx.x;  // B*L*H threads
  const int h = idx & (H_ - 1);
  const int row = idx >> 4;       // b*L + l
  const int b = row >> 10;
  const __hip_bfloat16* qr = qkv + (size_t)row * (3 * D_) + h * HD_;
  float q[HD_];
#pragma unroll
  for (int d = 0; d < HD_; ++d) q[d] = __bfloat162float(qr[d]);
  float s[C_];
  float mx = -1e30f;
#pragma unroll
  for (int c = 0; c < C_; ++c) {
    const float* kr = Ks + ((size_t)(b * C_ + c)) * D_ + h * HD_;
    float acc = 0.0f;
    for (int d = 0; d < HD_; ++d) acc += q[d] * kr[d];
    s[c] = acc * 0.125f;
    mx = fmaxf(mx, s[c]);
  }
  float sum = 0.0f;
#pragma unroll
  for (int c = 0; c < C_; ++c) { s[c] = __expf(s[c] - mx); sum += s[c]; }
  const float inv = 1.0f / sum;
  __hip_bfloat16* orow = o_bf + (size_t)row * D_ + h * HD_;
  for (int d = 0; d < HD_; ++d) {
    float acc = 0.0f;
#pragma unroll
    for (int c = 0; c < C_; ++c) acc += s[c] * Vs[((size_t)(b * C_ + c)) * D_ + h * HD_ + d];
    orow[d] = __float2bfloat16(acc * inv);
  }
}

// ---------------------------------------------------------------------------
// cond = LN1(hidden + att_final).  One block per row.
// ---------------------------------------------------------------------------
__global__ __launch_bounds__(256)
void final_ln_kernel(const float* __restrict__ hidden, const float* __restrict__ attf,
                     const float* __restrict__ g, const float* __restrict__ bt,
                     float* __restrict__ cond) {
  const int row = blockIdx.x, tid = threadIdx.x;
  __shared__ float xs[D_], red[32];
  float lsum = 0.0f;
#pragma unroll
  for (int k = 0; k < 4; ++k) {
    int d = tid + k * 256;
    float x = hidden[(size_t)row * D_ + d] + attf[(size_t)row * D_ + d];
    xs[d] = x;
    lsum += x;
  }
  float m = blockSum(lsum, red) * (1.0f / 1024.0f);
  float lv = 0.0f;
#pragma unroll
  for (int k = 0; k < 4; ++k) {
    int d = tid + k * 256;
    float dd = xs[d] - m;
    lv += dd * dd;
  }
  float var = blockSum(lv, red) * (1.0f / 1024.0f);
  float r = rsqrtf(var + 1e-5f);
#pragma unroll
  for (int k = 0; k < 4; ++k) {
    int d = tid + k * 256;
    cond[(size_t)row * D_ + d] = (xs[d] - m) * r * g[d] + bt[d];
  }
}

// ---------------------------------------------------------------------------
// Host orchestration
// ---------------------------------------------------------------------------
extern "C" void kernel_launch(void* const* d_in, const int* in_sizes, int n_in,
                              void* d_out, int out_size, void* d_ws, size_t ws_size,
                              hipStream_t stream) {
  const float* hidden = (const float*)d_in[0];
  const float* tpw    = (const float*)d_in[1];
  const float* tpb    = (const float*)d_in[2];
  const float* ln1g   = (const float*)d_in[3];
  const float* ln1b   = (const float*)d_in[4];
  const float* ln2g   = (const float*)d_in[5];
  const float* ln2b   = (const float*)d_in[6];
  const float* w_in   = (const float*)d_in[7];
  const float* b_in   = (const float*)d_in[8];
  const float* w_out  = (const float*)d_in[9];
  const float* b_out  = (const float*)d_in[10];
  const float* w1     = (const float*)d_in[11];
  const float* b1     = (const float*)d_in[12];
  const float* w2     = (const float*)d_in[13];
  const float* b2     = (const float*)d_in[14];
  const float* comp_q = (const float*)d_in[15];
  const float* comp_w = (const float*)d_in[16];
  const float* comp_b = (const float*)d_in[17];

  float* out      = (float*)d_out;
  float* cond_out = out;                             // (B,L,D)
  float* summ_out = out + (size_t)B_ * L_ * D_;      // (B,C,D)

  char* p = (char*)d_ws;
  auto take = [&](size_t bytes) -> char* {
    char* r = p;
    p += (bytes + 255) & ~(size_t)255;
    return r;
  };
  __hip_bfloat16* hid_bf   = (__hip_bfloat16*)take((size_t)B_ * L_ * D_ * 2);
  __hip_bfloat16* w_in_bf  = (__hip_bfloat16*)take((size_t)3 * D_ * D_ * 2);
  __hip_bfloat16* w_out_bf = (__hip_bfloat16*)take((size_t)D_ * D_ * 2);
  __hip_bfloat16* w1_bf    = (__hip_bfloat16*)take((size_t)2 * D_ * D_ * 2);
  __hip_bfloat16* w2_bf    = (__hip_bfloat16*)take((size_t)2 * D_ * D_ * 2);
  __hip_bfloat16* qkv_bf   = (__hip_bfloat16*)take((size_t)B_ * L_ * 3 * D_ * 2);
  __hip_bfloat16* o_bf     = (__hip_bfloat16*)take((size_t)B_ * L_ * D_ * 2);
  float* attf  = (float*)take((size_t)B_ * L_ * D_ * 4);
  float* tok_g = (float*)take((size_t)B_ * D_ * 4);
  float* buf_g = (float*)take((size_t)B_ * S_ * D_ * 4);
  float* kbuf  = (float*)take((size_t)B_ * S_ * D_ * 4);
  float* vbuf  = (float*)take((size_t)B_ * S_ * D_ * 4);
  float* Ks    = (float*)take((size_t)B_ * C_ * D_ * 4);
  float* Vs    = (float*)take((size_t)B_ * C_ * D_ * 4);

  auto cvt = [&](const float* src, __hip_bfloat16* dst, int n) {
    cvt_bf16_kernel<<<(n + 255) / 256, 256, 0, stream>>>(src, dst, n);
  };
  cvt(hidden, hid_bf, B_ * L_ * D_);
  cvt(w_in,  w_in_bf,  3 * D_ * D_);
  cvt(w_out, w_out_bf, D_ * D_);
  cvt(w1,    w1_bf,    2 * D_ * D_);
  cvt(w2,    w2_bf,    2 * D_ * D_);

  // ctx -> tok0 -> buf row 0 + its K/V
  init_tok_kernel<<<B_, 1024, 0, stream>>>(hidden, tpw, tpb, w_in_bf, b_in,
                                           tok_g, buf_g, kbuf, vbuf);

  // QKV projection of hidden (WMMA + TDM): (8192 x 1024) @ (3072 x 1024)^T -> bf16
  gemm_bf16_wmma<<<dim3(64, 24), 256, 0, stream>>>(hid_bf, w_in_bf, b_in,
                                                   nullptr, qkv_bf, D_, 3 * D_);

  // Sequential thought scan (one persistent block per batch)
  loop_kernel<<<B_, 1024, 0, stream>>>(qkv_bf, w_in_bf, b_in, w_out_bf, b_out,
                                       w1_bf, b1, w2_bf, b2,
                                       ln1g, ln1b, ln2g, ln2b,
                                       tok_g, buf_g, kbuf, vbuf);

  // Compressed summary + K/V of summ
  comp_kernel<<<B_ * C_, 1024, 0, stream>>>(comp_q, comp_w, comp_b, buf_g,
                                            w_in_bf, b_in, summ_out, Ks, Vs);

  // Final attention over the C=8 summary keys
  final_attn_kernel<<<(B_ * L_ * H_) / 256, 256, 0, stream>>>(qkv_bf, Ks, Vs, o_bf);

  // Output projection (WMMA + TDM): (8192 x 1024) @ (1024 x 1024)^T -> fp32
  gemm_bf16_wmma<<<dim3(64, 8), 256, 0, stream>>>(o_bf, w_out_bf, b_out,
                                                  attf, nullptr, D_, D_);

  // cond = LN1(hidden + att)
  final_ln_kernel<<<B_ * L_, 256, 0, stream>>>(hidden, attf, ln1g, ln1b, cond_out);
}
